// Dev_CharLSTM_13340168421545
// MI455X (gfx1250) — matrix operand order repaired and checked by reading, not compiled, not run on hardware
//
#include <hip/hip_runtime.h>
#include <hip/hip_bf16.h>

typedef _Float16 v8h  __attribute__((ext_vector_type(8)));
typedef _Float16 v16h __attribute__((ext_vector_type(16)));
typedef float    v8f  __attribute__((ext_vector_type(8)));

#define BATCH 512
#define SEQ   512
#define HS    256
#define EMB   8
#define VOCAB 80
#define NGATE (4*HS)          // 1024 gate columns
#define KPAD  288             // 256 (h) + 8 (x_e) + 24 zero pad, multiple of 32
#define AS    296             // LDS A-row stride in halfs (16B-aligned rows)
#define GSTRIDE (HS * KPAD)   // element offset between gate blocks in Bt (73728)

// ---------- one-time prep: Bt[n][k] = [U^T ; W^T ; 0] in fp16, K-contiguous ----------
__global__ void prep_bt(const float* __restrict__ U, const float* __restrict__ W,
                        _Float16* __restrict__ Bt) {
  int idx = blockIdx.x * blockDim.x + threadIdx.x;
  if (idx >= NGATE * KPAD) return;
  int n = idx / KPAD, k = idx - n * KPAD;
  float v = 0.f;
  if (k < HS)            v = U[(size_t)k * NGATE + n];        // U: (HS, 4HS)
  else if (k < HS + EMB) v = W[(size_t)(k - HS) * NGATE + n]; // W: (EMB, 4HS)
  Bt[idx] = (_Float16)v;
}

__global__ void prep_emb(const float* __restrict__ emb, _Float16* __restrict__ embH) {
  int idx = blockIdx.x * blockDim.x + threadIdx.x;
  if (idx < VOCAB * EMB) embH[idx] = (_Float16)emb[idx];
}

// ---------- fast, overflow-safe activations ----------
__device__ __forceinline__ float fsig(float x) {
  float e = __expf(-x);                 // x<<0 -> e=inf -> 0 ; x>>0 -> 1
  return 1.f / (1.f + e);
}
__device__ __forceinline__ float ftanh(float x) {
  float a = fabsf(x);
  float e = __expf(-2.f * a);           // a large -> e=0 -> t=1 (no overflow)
  float t = (1.f - e) / (1.f + e);
  return copysignf(t, x);
}

// ---------- persistent recurrent kernel: one WG per 16 batch rows ----------
__global__ void __launch_bounds__(512)
lstm_persist(const int* __restrict__ x, const _Float16* __restrict__ embH,
             const _Float16* __restrict__ Bt, float* __restrict__ hlast) {
  __shared__ _Float16 hA[16 * AS];      // A matrix: rows = batch tile, cols = K (h | x_e | 0)

  const int tid  = threadIdx.x;
  const int lane = tid & 31;
  const int wv   = tid >> 5;            // wave 0..15, owns hidden cols [16*wv, 16*wv+16)
  const int m    = lane & 15;           // A row / B column / C col index
  const int kh   = lane >> 4;           // K-half select
  const int b0   = blockIdx.x * 16;

  // zero the A buffer (h=0 at t=0; pad region stays 0 forever)
  for (int i = tid; i < 16 * AS; i += blockDim.x) hA[i] = (_Float16)0.f;
  __syncthreads();

  const int aBase = m * AS + 16 * kh;   // A per-lane base (halfs)

  // Single per-lane B base pointer (gate 0); gates 1..3 at +g*GSTRIDE elements
  // (folded into the 24-bit global_load immediate offset).
  const _Float16* pB = Bt + (size_t)(16 * wv + m) * KPAD + 16 * kh;

  v8f c = {0.f,0.f,0.f,0.f,0.f,0.f,0.f,0.f};   // cell state, lives in VGPRs
  v8f h = c;                                    // last hidden tile (fp32)

  const int* xrow = x + (size_t)(b0 + m) * SEQ; // used by wave 0 lanes 0..15

  // Opaque element offset (always 0 at runtime). Laundering the *offset* keeps
  // the load base rooted at the kernel-arg global pointer (-> global_load_b128,
  // pure LOADcnt), while LICM cannot hoist the loop-invariant B tiles (no spills).
  unsigned boff = 0;

  // Software-pipeline preload: K-block 0's four B tiles, carried across t.
  v16h bI = *(const v16h*)(pB + 0 * GSTRIDE);
  v16h bF = *(const v16h*)(pB + 1 * GSTRIDE);
  v16h bG = *(const v16h*)(pB + 2 * GSTRIDE);
  v16h bO = *(const v16h*)(pB + 3 * GSTRIDE);

  #pragma unroll 1
  for (int t = 0; t < SEQ; ++t) {
    asm volatile("" : "+v"(boff));      // opaque per-iteration; value stays 0
    const _Float16* pBt = pB + boff;

    // wave 0, lanes 0..15: gather this step's embeddings into A cols [256,264)
    if (wv == 0 && kh == 0) {
      int tok = xrow[t];
      v8h e = *(const v8h*)(embH + tok * EMB);
      *(v8h*)(&hA[m * AS + HS]) = e;
    }
    __syncthreads();                    // h stores (prev iter) + x_e visible

    v8f g0 = {0.f,0.f,0.f,0.f,0.f,0.f,0.f,0.f};
    v8f g1 = g0, g2 = g0, g3 = g0;
    #pragma unroll
    for (int kb = 0; kb < KPAD / 32; ++kb) {
      const int k0 = kb * 32;
      // next K-block to prefetch; at kb==8 wrap to 0 = next timestep's first
      // block (B is t-invariant), hiding L2 latency across the barrier phase.
      const int kn = (kb == KPAD / 32 - 1) ? 0 : k0 + 32;

      // A: 16x32 f16 tile, per-lane = two ds_read_b128
      v8h lo = *(const v8h*)(&hA[aBase + k0]);
      v8h hi = *(const v8h*)(&hA[aBase + k0 + 8]);
      v16h a;
      #pragma unroll
      for (int i = 0; i < 8; ++i) { a[i] = lo[i]; a[8 + i] = hi[i]; }

      // issue next B tiles before consuming current ones (double buffer)
      v16h nI = *(const v16h*)(pBt + 0 * GSTRIDE + kn);
      v16h nF = *(const v16h*)(pBt + 1 * GSTRIDE + kn);
      v16h nG = *(const v16h*)(pBt + 2 * GSTRIDE + kn);
      v16h nO = *(const v16h*)(pBt + 3 * GSTRIDE + kn);

      g0 = __builtin_amdgcn_wmma_f32_16x16x32_f16(false, a, false, bI, (short)0, g0, false, false);
      g1 = __builtin_amdgcn_wmma_f32_16x16x32_f16(false, a, false, bF, (short)0, g1, false, false);
      g2 = __builtin_amdgcn_wmma_f32_16x16x32_f16(false, a, false, bG, (short)0, g2, false, false);
      g3 = __builtin_amdgcn_wmma_f32_16x16x32_f16(false, a, false, bO, (short)0, g3, false, false);

      bI = nI; bF = nF; bG = nG; bO = nO;   // SSA rename (loop fully unrolled)
    }
    __syncthreads();                    // everyone done reading hA before overwrite

    // elementwise LSTM update; C/D layout: element (M = r + 8*kh, N = m)
    #pragma unroll
    for (int r = 0; r < 8; ++r) {
      float iv = fsig(g0[r]);
      float fv = fsig(g1[r]);
      float gv = ftanh(g2[r]);
      float ov = fsig(g3[r]);
      float cv = fv * c[r] + iv * gv;
      c[r] = cv;
      float hv = ov * ftanh(cv);
      h[r] = hv;
      hA[(r + 8 * kh) * AS + 16 * wv + m] = (_Float16)hv;  // next step's A
    }
    // next iteration's first __syncthreads orders these stores before reads
  }
  __syncthreads();

  // spill final hidden state (fp32) for the output projection
  #pragma unroll
  for (int r = 0; r < 8; ++r) {
    hlast[(size_t)(b0 + r + 8 * kh) * HS + 16 * wv + m] = h[r];
  }
}

// ---------- final projection: (512x256) @ (80x256)^T, 21 MFLOP ----------
__global__ void out_proj(const float* __restrict__ hlast, const float* __restrict__ Wout,
                         float* __restrict__ out) {
  int idx = blockIdx.x * blockDim.x + threadIdx.x;
  if (idx >= BATCH * VOCAB) return;
  int b = idx / VOCAB, v = idx - b * VOCAB;
  const float4* hp = (const float4*)(hlast + (size_t)b * HS);
  const float4* wp = (const float4*)(Wout + (size_t)v * HS);
  float s = 0.f;
  #pragma unroll 4
  for (int k = 0; k < HS / 4; ++k) {
    float4 a = hp[k], w = wp[k];
    s += a.x * w.x + a.y * w.y + a.z * w.z + a.w * w.w;
  }
  out[idx] = s;
}

extern "C" void kernel_launch(void* const* d_in, const int* in_sizes, int n_in,
                              void* d_out, int out_size, void* d_ws, size_t ws_size,
                              hipStream_t stream) {
  (void)in_sizes; (void)n_in; (void)out_size; (void)ws_size;
  const int*   x    = (const int*)  d_in[0];
  const float* emb  = (const float*)d_in[1];
  const float* W    = (const float*)d_in[2];
  const float* U    = (const float*)d_in[3];
  const float* Wout = (const float*)d_in[4];
  float* out = (float*)d_out;

  char* ws = (char*)d_ws;
  _Float16* Bt    = (_Float16*)ws;                       // 1024*288*2 = 589824 B
  _Float16* embH  = (_Float16*)(ws + 589824);            // 640*2      =   1280 B
  float*    hlast = (float*)   (ws + 589824 + 1280);     // 512*256*4  = 524288 B

  prep_bt  <<<(NGATE * KPAD + 255) / 256, 256, 0, stream>>>(U, W, Bt);
  prep_emb <<<(VOCAB * EMB + 255) / 256, 256, 0, stream>>>(emb, embH);
  lstm_persist<<<BATCH / 16, 512, 0, stream>>>(x, embH, Bt, hlast);
  out_proj <<<(BATCH * VOCAB + 255) / 256, 256, 0, stream>>>(hlast, Wout, out);
}
